// GCN_85933705658404
// MI455X (gfx1250) — matrix operand order repaired and checked by reading, not compiled, not run on hardware
//
#include <hip/hip_runtime.h>
#include <hip/hip_bf16.h>
#include <stdint.h>

typedef __attribute__((ext_vector_type(16))) _Float16 v16h;
typedef __attribute__((ext_vector_type(8)))  float    v8f;

#define N_NODES   100000
#define N_EDGES   1000000
#define D_IN      64
#define D_HID     64
#define N_CLASSES 32

// ---------------- elementwise helpers ----------------

__global__ void k_fill(float* __restrict__ p, float val, int n) {
  int i = blockIdx.x * blockDim.x + threadIdx.x;
  if (i < n) p[i] = val;
}

// deg accumulation over dst (self-loop handled by init value 1.0)
__global__ void k_deg_accum(const long long* __restrict__ ei, float* __restrict__ deg, int nE) {
  int i = blockIdx.x * blockDim.x + threadIdx.x;
  if (i < nE) {
    int d = (int)ei[(size_t)nE + i];
    atomicAdd(&deg[d], 1.0f);
  }
}

__global__ void k_deg_finish(float* __restrict__ dinv, int n) {
  int i = blockIdx.x * blockDim.x + threadIdx.x;
  if (i < n) dinv[i] = rsqrtf(dinv[i]);   // deg >= 1 always (self-loop)
}

// ---------------- weight packing into WMMA B-fragment layout ----------------
// B (K x N, 16-bit): lane l holds column n = l&15; half h = l>>4, element j ->
// K = 8h+j (j<8), K = 16+8h+(j-8) (j>=8) within the 32-K block.
// Fragment buffer index: ((ks*ntiles + nt)*32 + lane)*16 + j
__global__ void k_pack_w(const float* __restrict__ W, _Float16* __restrict__ Bp, int ncols) {
  int idx = blockIdx.x * blockDim.x + threadIdx.x;
  int ntiles = ncols >> 4;
  int total = 2 * ntiles * 32 * 16;       // ksteps(2) * ntiles * lanes * elems
  if (idx >= total) return;
  int j    = idx & 15;
  int lane = (idx >> 4) & 31;
  int nt   = (idx >> 9) % ntiles;
  int ks   = idx / (512 * ntiles);
  int half_ = lane >> 4;
  int n = nt * 16 + (lane & 15);
  int k = ks * 32 + half_ * 8 + (j < 8 ? j : 8 + j);
  Bp[idx] = (_Float16)W[k * ncols + n];
}

// ---------------- WMMA GEMM: C[N, ncols] = A[N, 64] * W ----------------
// One wave computes a 16-row x ncols strip. K = 64 -> 2 WMMA K-steps of 32.
__global__ void k_gemm_wmma(const float* __restrict__ A,
                            const _Float16* __restrict__ Bp,
                            float* __restrict__ C,
                            int nrows, int ncols) {
  int wave = blockIdx.x * (blockDim.x >> 5) + (threadIdx.x >> 5);
  int lane = threadIdx.x & 31;
  int m0 = wave << 4;
  if (m0 >= nrows) return;                // wave-uniform guard: EXEC stays all-1s
  int half_ = lane >> 4;
  int r     = lane & 15;

  // A fragments (16-bit A 16x32 layout): lane holds row m0+r;
  // lanes 0-15: K = {h*8+0..7, h*8+16..23} with h=0; lanes 16-31: h=1.
  const float* arow = A + (size_t)(m0 + r) * 64;
  v16h a0, a1;
  #pragma unroll
  for (int j = 0; j < 8; ++j) {
    a0[j]     = (_Float16)arow[half_ * 8 + j];
    a0[8 + j] = (_Float16)arow[half_ * 8 + 16 + j];
    a1[j]     = (_Float16)arow[32 + half_ * 8 + j];
    a1[8 + j] = (_Float16)arow[32 + half_ * 8 + 16 + j];
  }

  int ntiles = ncols >> 4;
  for (int nt = 0; nt < ntiles; ++nt) {
    v8f c = {};
    v16h b0 = *(const v16h*)(Bp + ((size_t)(0 * ntiles + nt) * 32 + lane) * 16);
    c = __builtin_amdgcn_wmma_f32_16x16x32_f16(false, a0, false, b0, (short)0, c, false, false);
    v16h b1 = *(const v16h*)(Bp + ((size_t)(1 * ntiles + nt) * 32 + lane) * 16);
    c = __builtin_amdgcn_wmma_f32_16x16x32_f16(false, a1, false, b1, (short)0, c, false, false);
    // C/D layout: lane -> col nt*16 + (lane&15); VGPR vr -> row m0 + vr + half*8
    float* cb = C + (size_t)(m0 + half_ * 8) * ncols + nt * 16 + r;
    #pragma unroll
    for (int vr = 0; vr < 8; ++vr) cb[(size_t)vr * ncols] = c[vr];
  }
}

// ---------------- scatter-add aggregation (L2-resident atomics) ----------------
// One wave32 per edge. Edge id forced uniform -> index/dinv loads scalarize to
// s_load (KMcnt), issued once per wave instead of once per lane. Each lane moves
// a float2 (F=64) or a float (F=32). Virtual edges e in [nE, nE+nN) = self-loops.
__global__ void k_aggregate(const float* __restrict__ msg,
                            const long long* __restrict__ ei,
                            const float* __restrict__ dinv,
                            float* __restrict__ agg,
                            int nE, int nN, int F) {
  int wave = blockIdx.x * (blockDim.x >> 5) + (threadIdx.x >> 5);
  int lane = threadIdx.x & 31;
  if (wave >= nE + nN) return;            // wave-uniform guard
  int e = __builtin_amdgcn_readfirstlane(wave);
  int s, d;
  if (e < nE) {
    s = (int)ei[e];
    d = (int)ei[(size_t)nE + e];
  } else {
    s = d = e - nE;
  }
  s = __builtin_amdgcn_readfirstlane(s);
  d = __builtin_amdgcn_readfirstlane(d);
  float w = dinv[s] * dinv[d];
  if (F == 64) {
    float2 v = *(const float2*)(msg + (size_t)s * 64 + lane * 2);
    float* a = agg + (size_t)d * 64 + lane * 2;
    atomicAdd(a,     v.x * w);
    atomicAdd(a + 1, v.y * w);
  } else {                                // F == 32
    float v = msg[(size_t)s * 32 + lane];
    atomicAdd(agg + (size_t)d * 32 + lane, v * w);
  }
}

__global__ void k_bias_relu(float* __restrict__ h, const float* __restrict__ b,
                            int n, int logF) {
  int idx = blockIdx.x * blockDim.x + threadIdx.x;
  int total = n << logF;
  if (idx >= total) return;
  float v = h[idx] + b[idx & ((1 << logF) - 1)];
  h[idx] = v > 0.0f ? v : 0.0f;
}

// ---------------- final bias + log_softmax: one wave32 per node, lane=class ----
__global__ void k_logsoftmax32(const float* __restrict__ agg, const float* __restrict__ b,
                               float* __restrict__ out, int n) {
  int wave = blockIdx.x * (blockDim.x >> 5) + (threadIdx.x >> 5);
  int lane = threadIdx.x & 31;
  if (wave >= n) return;
  float v = agg[(size_t)wave * 32 + lane] + b[lane];
  float m = v;
  #pragma unroll
  for (int off = 16; off > 0; off >>= 1) m = fmaxf(m, __shfl_xor(m, off, 32));
  float ex = __expf(v - m);
  float s = ex;
  #pragma unroll
  for (int off = 16; off > 0; off >>= 1) s += __shfl_xor(s, off, 32);
  out[(size_t)wave * 32 + lane] = (v - m) - __logf(s);
}

// ---------------- host orchestration ----------------

extern "C" void kernel_launch(void* const* d_in, const int* in_sizes, int n_in,
                              void* d_out, int out_size, void* d_ws, size_t ws_size,
                              hipStream_t stream) {
  (void)in_sizes; (void)n_in; (void)out_size; (void)ws_size;

  const float*     x  = (const float*)d_in[0];
  const long long* ei = (const long long*)d_in[1];
  const float* W[4] = {(const float*)d_in[2], (const float*)d_in[4],
                       (const float*)d_in[6], (const float*)d_in[8]};
  const float* B[4] = {(const float*)d_in[3], (const float*)d_in[5],
                       (const float*)d_in[7], (const float*)d_in[9]};
  float* out = (float*)d_out;

  // workspace layout (256B aligned slabs)
  char* ws = (char*)d_ws;
  size_t off = 0;
  auto alloc = [&](size_t bytes) -> void* {
    void* p = ws + off;
    off = (off + bytes + 255) & ~(size_t)255;
    return p;
  };
  float* dinv = (float*)alloc((size_t)N_NODES * sizeof(float));
  int ncols[4] = {D_HID, D_HID, D_HID, N_CLASSES};
  _Float16* Bp[4];
  for (int l = 0; l < 4; ++l)
    Bp[l] = (_Float16*)alloc((size_t)2 * (ncols[l] / 16) * 32 * 16 * sizeof(_Float16));
  float* bufA = (float*)alloc((size_t)N_NODES * 64 * sizeof(float));
  float* bufB = (float*)alloc((size_t)N_NODES * 64 * sizeof(float));

  const int T = 256;

  // degree -> dinv = rsqrt(deg) (self-loop included via init 1.0)
  k_fill<<<(N_NODES + T - 1) / T, T, 0, stream>>>(dinv, 1.0f, N_NODES);
  k_deg_accum<<<(N_EDGES + T - 1) / T, T, 0, stream>>>(ei, dinv, N_EDGES);
  k_deg_finish<<<(N_NODES + T - 1) / T, T, 0, stream>>>(dinv, N_NODES);

  // pack weights into WMMA B-fragment layout (f16)
  for (int l = 0; l < 4; ++l) {
    int total = 2 * (ncols[l] / 16) * 512;
    k_pack_w<<<(total + T - 1) / T, T, 0, stream>>>(W[l], Bp[l], ncols[l]);
  }

  const float* hin = x;
  for (int l = 0; l < 4; ++l) {
    int F = ncols[l];
    int logF = (F == 64) ? 6 : 5;

    // dense projection via WMMA: bufB = hin @ W[l]
    int waves  = (N_NODES + 15) / 16;
    int blocks = (waves + 7) / 8;        // 8 wave32 per 256-thread block
    k_gemm_wmma<<<blocks, T, 0, stream>>>(hin, Bp[l], bufB, N_NODES, F);

    // aggregate: bufA = scatter_add(norm * gather(bufB)), one wave per edge
    int nAgg = N_NODES * F;
    k_fill<<<(nAgg + T - 1) / T, T, 0, stream>>>(bufA, 0.0f, nAgg);
    int nMsgWaves = N_EDGES + N_NODES;
    k_aggregate<<<(nMsgWaves + 7) / 8, T, 0, stream>>>(bufB, ei, dinv, bufA,
                                                       N_EDGES, N_NODES, F);
    if (l < 3) {
      k_bias_relu<<<(nAgg + T - 1) / T, T, 0, stream>>>(bufA, B[l], N_NODES, logF);
      hin = bufA;
    } else {
      k_logsoftmax32<<<(N_NODES + 7) / 8, T, 0, stream>>>(bufA, B[l], out, N_NODES);
    }
  }
}
